// MultiHeadAttention_55972013802146
// MI455X (gfx1250) — matrix operand order repaired and checked by reading, not compiled, not run on hardware
//
#include <hip/hip_runtime.h>

// MHA: B=4, S=2048, E=1024, H=16, D=64.
// d_out = [out: B*S*E fp32][attn: B*H*S*S fp32]
// d_ws  = [Qh bf16 B*H*S*D][Kh bf16 B*H*S*D][Vt bf16 B*H*D*S]; ctx reuses Qh.
// ws bytes needed: 3 * 8,388,608 * 2 = 50,331,648.

#define Bc 4
#define Sc 2048
#define Ec 1024
#define Hc 16
#define Dc 64

typedef __attribute__((ext_vector_type(16))) __bf16 v16bf;
typedef __attribute__((ext_vector_type(8)))  __bf16 v8bf;
typedef __attribute__((ext_vector_type(8)))  float  v8f;
typedef __attribute__((ext_vector_type(4)))  float  v4f;

static __device__ __forceinline__ v8f wmma_bf16(v16bf a, v16bf b, v8f c) {
  return __builtin_amdgcn_wmma_f32_16x16x32_bf16(false, a, false, b, (short)0, c,
                                                 false, false);
}

// ---- A fragment: 16x32, source row-major [rows=M, contiguous K] ----
// lane = half*16 + m; elems 0..7 <- K=half*8+e ; elems 8..15 <- K=16+half*8+(e-8)
static __device__ __forceinline__ v16bf load_a_bf16(const __bf16* base, int ld,
                                                    int row0, int k0, int lane) {
  const int half = lane >> 4, m = lane & 15;
  const __bf16* p = base + (size_t)(row0 + m) * ld + k0 + half * 8;
  union { v16bf v; v8bf h[2]; } u;
  u.h[0] = *(const v8bf*)p;
  u.h[1] = *(const v8bf*)(p + 16);
  return u.v;
}

static __device__ __forceinline__ v16bf load_a_f32(const float* base, int ld,
                                                   int row0, int k0, int lane) {
  const int half = lane >> 4, m = lane & 15;
  const float* p = base + (size_t)(row0 + m) * ld + k0 + half * 8;
  v4f f0 = *(const v4f*)p;
  v4f f1 = *(const v4f*)(p + 4);
  v4f f2 = *(const v4f*)(p + 16);
  v4f f3 = *(const v4f*)(p + 20);
  v16bf r;
#pragma unroll
  for (int i = 0; i < 4; ++i) {
    r[i]      = (__bf16)f0[i];
    r[4 + i]  = (__bf16)f1[i];
    r[8 + i]  = (__bf16)f2[i];
    r[12 + i] = (__bf16)f3[i];
  }
  return r;
}

// ---- B fragment: 32x16, source = row-major B^T [rows=N, contiguous K] ----
// lane = half*16 + n; elems 0..7 <- K=half*8+e ; elems 8..15 <- K=16+half*8+(e-8)
static __device__ __forceinline__ v16bf load_b_bf16(const __bf16* base, int ld,
                                                    int row0, int k0, int lane) {
  const int half = lane >> 4, n = lane & 15;
  const __bf16* p = base + (size_t)(row0 + n) * ld + k0 + half * 8;
  union { v16bf v; v8bf h[2]; } u;
  u.h[0] = *(const v8bf*)p;
  u.h[1] = *(const v8bf*)(p + 16);
  return u.v;
}

static __device__ __forceinline__ v16bf load_b_f32(const float* base, int ld,
                                                   int row0, int k0, int lane) {
  const int half = lane >> 4, n = lane & 15;
  const float* p = base + (size_t)(row0 + n) * ld + k0 + half * 8;
  v4f f0 = *(const v4f*)p;
  v4f f1 = *(const v4f*)(p + 4);
  v4f f2 = *(const v4f*)(p + 16);
  v4f f3 = *(const v4f*)(p + 20);
  v16bf r;
#pragma unroll
  for (int i = 0; i < 4; ++i) {
    r[i]      = (__bf16)f0[i];
    r[4 + i]  = (__bf16)f1[i];
    r[8 + i]  = (__bf16)f2[i];
    r[12 + i] = (__bf16)f3[i];
  }
  return r;
}

// =====================================================================
// Projections: X[8192,1024]f32 x W[1024,1024]f32^T + b -> bf16 head layouts.
// Block 256 (8 waves, 2Mx4N), block tile 128x128, wave tile 64x32 (4x2 WMMA).
// grid (64, 8). MODE: 0 -> [B,H,S,D] (Q/K), 1 -> [B,H,D,S] (V transposed)
// =====================================================================
template <int MODE>
__global__ void __launch_bounds__(256)
proj_kernel(const float* __restrict__ X, const float* __restrict__ W,
            const float* __restrict__ bias, __bf16* __restrict__ outp) {
  const int lane = threadIdx.x & 31;
  const int w = threadIdx.x >> 5;
  const int wm = w & 1, wn = w >> 1;
  const int m0 = blockIdx.x * 128 + wm * 64;
  const int n0 = blockIdx.y * 128 + wn * 32;

  v8f acc[4][2] = {};
  for (int k0 = 0; k0 < Ec; k0 += 32) {
    // prefetch two k-steps ahead (speculative; dropped if OOB)
    __builtin_prefetch(X + (size_t)(m0 + (lane & 15)) * Ec + k0 + 64, 0, 1);
    __builtin_prefetch(W + (size_t)(n0 + (lane & 15)) * Ec + k0 + 64, 0, 1);
    v16bf b0 = load_b_f32(W, Ec, n0, k0, lane);
    v16bf b1 = load_b_f32(W, Ec, n0 + 16, k0, lane);
    v16bf a[4];
#pragma unroll
    for (int i = 0; i < 4; ++i) a[i] = load_a_f32(X, Ec, m0 + 16 * i, k0, lane);
#pragma unroll
    for (int i = 0; i < 4; ++i) {
      acc[i][0] = wmma_bf16(a[i], b0, acc[i][0]);
      acc[i][1] = wmma_bf16(a[i], b1, acc[i][1]);
    }
  }

  const int half = lane >> 4, nl = lane & 15;
#pragma unroll
  for (int i = 0; i < 4; ++i)
#pragma unroll
    for (int j = 0; j < 2; ++j) {
      const int n = n0 + j * 16 + nl;
      const float bv = bias[n];
      const int h = n >> 6, d = n & 63;
#pragma unroll
      for (int r = 0; r < 8; ++r) {
        const int m = m0 + i * 16 + r + 8 * half;  // 0..8191
        const int b = m >> 11, s = m & (Sc - 1);
        const float v = acc[i][j][r] + bv;
        if (MODE == 0) {
          outp[((((size_t)(b * Hc + h)) * Sc + s) << 6) + d] = (__bf16)v;
        } else {
          outp[(((size_t)(b * Hc + h)) * Dc + d) * Sc + s] = (__bf16)v;
        }
      }
    }
}

// =====================================================================
// scores = Q K^T / 8 : per (b,h) [2048,64] x [2048,64]^T -> fp32 [2048,2048]
// Block tile 128x128 (waves 2Mx4N, wave tile 64x32). grid (16, 16, 64)
// =====================================================================
__global__ void __launch_bounds__(256)
scores_kernel(const __bf16* __restrict__ Qh, const __bf16* __restrict__ Kh,
              float* __restrict__ attn) {
  const int lane = threadIdx.x & 31;
  const int w = threadIdx.x >> 5;
  const int wm = w & 1, wn = w >> 1;
  const int bh = blockIdx.z;
  const int m0 = blockIdx.x * 128 + wm * 64;
  const int n0 = blockIdx.y * 128 + wn * 32;
  const __bf16* Qp = Qh + (size_t)bh * Sc * Dc;
  const __bf16* Kp = Kh + (size_t)bh * Sc * Dc;
  float* Ap = attn + (size_t)bh * Sc * Sc;

  v8f acc[4][2] = {};
#pragma unroll
  for (int k0 = 0; k0 < Dc; k0 += 32) {
    v16bf b0 = load_b_bf16(Kp, Dc, n0, k0, lane);
    v16bf b1 = load_b_bf16(Kp, Dc, n0 + 16, k0, lane);
    v16bf a[4];
#pragma unroll
    for (int i = 0; i < 4; ++i) a[i] = load_a_bf16(Qp, Dc, m0 + 16 * i, k0, lane);
#pragma unroll
    for (int i = 0; i < 4; ++i) {
      acc[i][0] = wmma_bf16(a[i], b0, acc[i][0]);
      acc[i][1] = wmma_bf16(a[i], b1, acc[i][1]);
    }
  }

  const int half = lane >> 4, nl = lane & 15;
#pragma unroll
  for (int i = 0; i < 4; ++i)
#pragma unroll
    for (int j = 0; j < 2; ++j) {
      const int n = n0 + j * 16 + nl;
#pragma unroll
      for (int r = 0; r < 8; ++r) {
        const int m = m0 + i * 16 + r + 8 * half;
        Ap[(size_t)m * Sc + n] = acc[i][j][r] * 0.125f;
      }
    }
}

// =====================================================================
// softmax in place over last dim (2048). One wave per row, 64 elems/lane.
// grid 16384 blocks x 256
// =====================================================================
__global__ void __launch_bounds__(256)
softmax_kernel(float* __restrict__ attn) {
  const int lane = threadIdx.x & 31;
  const int w = threadIdx.x >> 5;
  const size_t row = (size_t)blockIdx.x * 8 + w;
  float* p = attn + row * Sc;

  float v[64];
#pragma unroll
  for (int i = 0; i < 16; ++i) {
    v4f t = *(const v4f*)(p + i * 128 + lane * 4);
    v[4 * i + 0] = t[0]; v[4 * i + 1] = t[1];
    v[4 * i + 2] = t[2]; v[4 * i + 3] = t[3];
  }
  float mx = -3.402823466e38f;
#pragma unroll
  for (int i = 0; i < 64; ++i) mx = fmaxf(mx, v[i]);
#pragma unroll
  for (int off = 16; off > 0; off >>= 1) mx = fmaxf(mx, __shfl_xor(mx, off, 32));

  float sum = 0.0f;
#pragma unroll
  for (int i = 0; i < 64; ++i) { v[i] = __expf(v[i] - mx); sum += v[i]; }
#pragma unroll
  for (int off = 16; off > 0; off >>= 1) sum += __shfl_xor(sum, off, 32);

  const float inv = 1.0f / sum;
#pragma unroll
  for (int i = 0; i < 16; ++i) {
    v4f t;
    t[0] = v[4 * i + 0] * inv; t[1] = v[4 * i + 1] * inv;
    t[2] = v[4 * i + 2] * inv; t[3] = v[4 * i + 3] * inv;
    *(v4f*)(p + i * 128 + lane * 4) = t;
  }
}

// =====================================================================
// ctx = attn x V : per (b,h) [2048,2048]f32 x Vt[64,2048]bf16 -> ctx bf16 [B*S,E]
// Block tile 256x64 (waves 4Mx2N, wave tile 64x32). grid (8, 1, 64)
// =====================================================================
__global__ void __launch_bounds__(256)
ctx_kernel(const float* __restrict__ attn, const __bf16* __restrict__ Vt,
           __bf16* __restrict__ ctx) {
  const int lane = threadIdx.x & 31;
  const int w = threadIdx.x >> 5;
  const int wm = w & 3, wn = w >> 2;
  const int bh = blockIdx.z;
  const int bb = bh >> 4, h = bh & 15;
  const int m0 = blockIdx.x * 256 + wm * 64;
  const int n0 = wn * 32;
  const float* Ap = attn + (size_t)bh * Sc * Sc;
  const __bf16* Vp = Vt + (size_t)bh * Dc * Sc;

  v8f acc[4][2] = {};
  for (int k0 = 0; k0 < Sc; k0 += 32) {
    __builtin_prefetch(Ap + (size_t)(m0 + (lane & 15)) * Sc + k0 + 64, 0, 1);
    v16bf b0 = load_b_bf16(Vp, Sc, n0, k0, lane);
    v16bf b1 = load_b_bf16(Vp, Sc, n0 + 16, k0, lane);
    v16bf a[4];
#pragma unroll
    for (int i = 0; i < 4; ++i) a[i] = load_a_f32(Ap, Sc, m0 + 16 * i, k0, lane);
#pragma unroll
    for (int i = 0; i < 4; ++i) {
      acc[i][0] = wmma_bf16(a[i], b0, acc[i][0]);
      acc[i][1] = wmma_bf16(a[i], b1, acc[i][1]);
    }
  }

  const int half = lane >> 4, nl = lane & 15;
#pragma unroll
  for (int i = 0; i < 4; ++i)
#pragma unroll
    for (int j = 0; j < 2; ++j) {
      const int n = n0 + j * 16 + nl;  // 0..63
#pragma unroll
      for (int r = 0; r < 8; ++r) {
        const int s = m0 + i * 16 + r + 8 * half;
        const size_t idx = ((size_t)(bb * Sc + s)) * Ec + h * Dc + n;
        ctx[idx] = (__bf16)acc[i][j][r];
      }
    }
}

// =====================================================================
// out = ctx x Wo^T + bo : [8192,1024]bf16 x Wo[1024,1024]f32 -> fp32
// Block tile 128x128 (waves 2Mx4N, wave tile 64x32). grid (64, 8)
// =====================================================================
__global__ void __launch_bounds__(256)
out_kernel(const __bf16* __restrict__ ctx, const float* __restrict__ Wo,
           const float* __restrict__ bo, float* __restrict__ out) {
  const int lane = threadIdx.x & 31;
  const int w = threadIdx.x >> 5;
  const int wm = w & 1, wn = w >> 1;
  const int m0 = blockIdx.x * 128 + wm * 64;
  const int n0 = blockIdx.y * 128 + wn * 32;

  v8f acc[4][2] = {};
  for (int k0 = 0; k0 < Ec; k0 += 32) {
    __builtin_prefetch(ctx + (size_t)(m0 + (lane & 15)) * Ec + k0 + 64, 0, 1);
    __builtin_prefetch(Wo + (size_t)(n0 + (lane & 15)) * Ec + k0 + 64, 0, 1);
    v16bf b0 = load_b_f32(Wo, Ec, n0, k0, lane);
    v16bf b1 = load_b_f32(Wo, Ec, n0 + 16, k0, lane);
    v16bf a[4];
#pragma unroll
    for (int i = 0; i < 4; ++i) a[i] = load_a_bf16(ctx, Ec, m0 + 16 * i, k0, lane);
#pragma unroll
    for (int i = 0; i < 4; ++i) {
      acc[i][0] = wmma_bf16(a[i], b0, acc[i][0]);
      acc[i][1] = wmma_bf16(a[i], b1, acc[i][1]);
    }
  }

  const int half = lane >> 4, nl = lane & 15;
#pragma unroll
  for (int i = 0; i < 4; ++i)
#pragma unroll
    for (int j = 0; j < 2; ++j) {
      const int n = n0 + j * 16 + nl;
      const float bv = bo[n];
#pragma unroll
      for (int r = 0; r < 8; ++r) {
        const int m = m0 + i * 16 + r + 8 * half;
        out[(size_t)m * Ec + n] = acc[i][j][r] + bv;
      }
    }
}

extern "C" void kernel_launch(void* const* d_in, const int* in_sizes, int n_in,
                              void* d_out, int out_size, void* d_ws, size_t ws_size,
                              hipStream_t stream) {
  const float* query  = (const float*)d_in[0];
  const float* key_in = (const float*)d_in[1];
  const float* value  = (const float*)d_in[2];
  const float* Wq = (const float*)d_in[3];
  const float* bq = (const float*)d_in[4];
  const float* Wk = (const float*)d_in[5];
  const float* bk = (const float*)d_in[6];
  const float* Wv = (const float*)d_in[7];
  const float* bv = (const float*)d_in[8];
  const float* Wo = (const float*)d_in[9];
  const float* bo = (const float*)d_in[10];

  float* out  = (float*)d_out;
  float* attn = out + (size_t)Bc * Sc * Ec;  // out first, attn second

  const size_t headElems = (size_t)Bc * Hc * Sc * Dc;  // 8,388,608
  __bf16* Qh  = (__bf16*)d_ws;
  __bf16* Kh  = Qh + headElems;
  __bf16* Vt  = Kh + headElems;
  __bf16* ctx = Qh;  // Qh is dead after scores_kernel; reuse for ctx

  dim3 blk(256);
  dim3 gproj(64, 8);           // 8192/128 x 1024/128
  proj_kernel<0><<<gproj, blk, 0, stream>>>(query,  Wq, bq, Qh);
  proj_kernel<0><<<gproj, blk, 0, stream>>>(key_in, Wk, bk, Kh);
  proj_kernel<1><<<gproj, blk, 0, stream>>>(value,  Wv, bv, Vt);

  dim3 gsc(16, 16, Bc * Hc);   // 2048/128 x 2048/128 x 64
  scores_kernel<<<gsc, blk, 0, stream>>>(Qh, Kh, attn);

  softmax_kernel<<<dim3(16384), blk, 0, stream>>>(attn);  // 131072 rows / 8

  dim3 gctx(8, 1, Bc * Hc);    // 2048/256
  ctx_kernel<<<gctx, blk, 0, stream>>>(attn, Vt, ctx);

  out_kernel<<<gproj, blk, 0, stream>>>(ctx, Wo, bo, out);
}
